// QLlamaAttention_78855599555416
// MI455X (gfx1250) — matrix operand order repaired and checked by reading, not compiled, not run on hardware
//
#include <hip/hip_runtime.h>

// ---------------------------------------------------------------------------
// Types
// ---------------------------------------------------------------------------
typedef __bf16 bf16_t;
typedef __attribute__((ext_vector_type(16))) __bf16 v16bf;
typedef __attribute__((ext_vector_type(8)))  __bf16 v8bf;
typedef __attribute__((ext_vector_type(8)))  float  v8f;
typedef __attribute__((ext_vector_type(4)))  int    v4i;

// Problem constants (from reference)
#define BATCH 2
#define SEQ   1024
#define HID   4096
#define NHEAD 32
#define HDIM  128
#define MROWS (BATCH*SEQ)        // 2048
#define GROUPS_PER_ROW (HID/32)  // 128

// ---------------------------------------------------------------------------
// bf16 helpers (round-to-nearest-even truncation)
// ---------------------------------------------------------------------------
__device__ inline bf16_t f2bf(float f) {
    unsigned u = __builtin_bit_cast(unsigned, f);
    unsigned r = (u + 0x7FFFu + ((u >> 16) & 1u)) >> 16;
    return __builtin_bit_cast(bf16_t, (unsigned short)r);
}
__device__ inline unsigned short f2bf_bits(float f) {
    unsigned u = __builtin_bit_cast(unsigned, f);
    return (unsigned short)((u + 0x7FFFu + ((u >> 16) & 1u)) >> 16);
}
__device__ inline float bf2f(bf16_t b) {
    unsigned u = ((unsigned)__builtin_bit_cast(unsigned short, b)) << 16;
    return __builtin_bit_cast(float, u);
}

// 16 contiguous bf16 -> v16bf (two b128 loads)
__device__ inline v16bf ld16(const bf16_t* p) {
    const v8bf* q = (const v8bf*)p;
    v8bf x = q[0], y = q[1];
    return __builtin_shufflevector(x, y, 0,1,2,3,4,5,6,7,8,9,10,11,12,13,14,15);
}
// A-matrix style fragment: 8 bf16 at p0, 8 bf16 at p1
__device__ inline v16bf ld_split(const bf16_t* p0, const bf16_t* p1) {
    v8bf x = *(const v8bf*)p0;
    v8bf y = *(const v8bf*)p1;
    return __builtin_shufflevector(x, y, 0,1,2,3,4,5,6,7,8,9,10,11,12,13,14,15);
}

// ---------------------------------------------------------------------------
// CDNA5 async copy: global -> LDS, 16 bytes per lane, tracked by ASYNCcnt.
// Toolchain declares the builtin with int4* parameters (probe: round-2 diag).
// Falls back to a direct LDS store if the builtin is not declared.
// ---------------------------------------------------------------------------
#if __has_builtin(__builtin_amdgcn_global_load_async_to_lds_b128)
#define HAVE_ASYNC_LDS 1
#else
#define HAVE_ASYNC_LDS 0
#endif

__device__ inline void async_cp16(bf16_t* lds_dst, const bf16_t* gsrc) {
#if HAVE_ASYNC_LDS
    __builtin_amdgcn_global_load_async_to_lds_b128(
        (v4i*)gsrc, (v4i*)lds_dst, 0, 0);
#else
    *(uint4*)lds_dst = *(const uint4*)gsrc;
#endif
}

#if __has_builtin(__builtin_amdgcn_s_wait_asynccnt)
#define WAIT_ASYNC(n) __builtin_amdgcn_s_wait_asynccnt(n)
#elif HAVE_ASYNC_LDS
#define WAIT_ASYNC(n) asm volatile("s_wait_asynccnt %0" :: "i"(n) : "memory")
#else
#define WAIT_ASYNC(n) ((void)0)
#endif

// ---------------------------------------------------------------------------
// Kernel 1: mixed 4/6/8-bit fake-quant (group=32 along K) -> bf16 dequantized
// Columns [0,2048)=4b, [2048,3072)=6b, [3072,4096)=8b (reference P4/P6/P8).
// ---------------------------------------------------------------------------
__global__ __launch_bounds__(256) void fakequant_kernel(
    const float* __restrict__ src, bf16_t* __restrict__ dst, long long ngroups)
{
    long long g = (long long)blockIdx.x * blockDim.x + threadIdx.x;
    if (g >= ngroups) return;
    int cg   = (int)(g & (GROUPS_PER_ROW - 1));
    int col0 = cg * 32;
    float maxq = (col0 < 2048) ? 7.0f : (col0 < 3072 ? 31.0f : 127.0f);

    const float4* p4 = (const float4*)(src + g * 32);
    float v[32];
    float amax = 0.0f;
#pragma unroll
    for (int i = 0; i < 8; ++i) {
        float4 t = p4[i];
        v[i*4+0] = t.x; v[i*4+1] = t.y; v[i*4+2] = t.z; v[i*4+3] = t.w;
        amax = fmaxf(amax, fmaxf(fmaxf(fabsf(t.x), fabsf(t.y)),
                                 fmaxf(fabsf(t.z), fabsf(t.w))));
    }
    float s = amax / maxq;
    if (s == 0.0f) s = 1.0f;
    float inv_s = 1.0f / s;
#pragma unroll
    for (int i = 0; i < 32; ++i) {
        float q = rintf(v[i] * inv_s);
        q = fminf(fmaxf(q, -maxq - 1.0f), maxq);
        v[i] = q * s;
    }
    unsigned* out = (unsigned*)(dst + g * 32);
#pragma unroll
    for (int i = 0; i < 16; ++i)
        out[i] = (unsigned)f2bf_bits(v[2*i]) | ((unsigned)f2bf_bits(v[2*i+1]) << 16);
}

// ---------------------------------------------------------------------------
// Kernel 2: bf16 WMMA GEMM  C[M,N] = A[M,K] * B[N,K]^T
// Block: 256 threads = 8 waves, tile BM=128 x BN=128, BK=32, double-buffered
// LDS filled by async global->LDS copies (ASYNCcnt-pipelined).
// Wave (wm,wn) computes a 32x64 patch = 2x4 WMMA tiles (8 WMMAs / K-step).
// ---------------------------------------------------------------------------
template <bool OUT_BF16>
__global__ __launch_bounds__(256) void gemm_bf16(
    const bf16_t* __restrict__ A, const bf16_t* __restrict__ B,
    void* __restrict__ Cout, int M, int N, int K)
{
    __shared__ bf16_t As[2][128 * 32];
    __shared__ bf16_t Bs[2][128 * 32];

    const int tid  = threadIdx.x;
    const int lane = tid & 31;
    const int wave = tid >> 5;
    const int wm   = wave & 3;       // 0..3  (rows, 32 each)
    const int wn   = wave >> 2;      // 0..1  (cols, 64 each)
    const int half = lane >> 4;      // 0/1
    const int l15  = lane & 15;

    const int nBase = blockIdx.x * 128;
    const int mBase = blockIdx.y * 128;

    // tile-load mapping: 128 rows x 32 cols bf16 = 8 KB; 32 B per thread
    const int trow = tid >> 1;
    const int tcol = (tid & 1) * 16;
    const bf16_t* aSrc = A + (size_t)(mBase + trow) * K + tcol;
    const bf16_t* bSrc = B + (size_t)(nBase + trow) * K + tcol;

    auto preload = [&](int buf, int k0) {
        bf16_t* ad = &As[buf][trow * 32 + tcol];
        bf16_t* bd = &Bs[buf][trow * 32 + tcol];
        async_cp16(ad,     aSrc + k0);
        async_cp16(ad + 8, aSrc + k0 + 8);
        async_cp16(bd,     bSrc + k0);
        async_cp16(bd + 8, bSrc + k0 + 8);
    };

    v8f acc[2][4] = {};

    preload(0, 0);
    int buf = 0;
    for (int k0 = 0; k0 < K; k0 += 32) {
        const bool has_next = (k0 + 32) < K;
        if (has_next) preload(buf ^ 1, k0 + 32);        // prefetch next tile
        if (has_next) { WAIT_ASYNC(4); } else { WAIT_ASYNC(0); }  // current tile landed
        __syncthreads();

        v16bf af[2], bfv[4];
#pragma unroll
        for (int mi = 0; mi < 2; ++mi) {
            const bf16_t* base = &As[buf][(wm * 32 + mi * 16 + l15) * 32];
            af[mi] = ld16(base + half * 16);
        }
#pragma unroll
        for (int ni = 0; ni < 4; ++ni) {
            const bf16_t* base = &Bs[buf][(wn * 64 + ni * 16 + l15) * 32];
            bfv[ni] = ld16(base + half * 16);
        }
#pragma unroll
        for (int mi = 0; mi < 2; ++mi)
#pragma unroll
            for (int ni = 0; ni < 4; ++ni)
                acc[mi][ni] = __builtin_amdgcn_wmma_f32_16x16x32_bf16(
                    false, af[mi], false, bfv[ni], (short)0, acc[mi][ni], false, false);

        __syncthreads();   // protect this buffer before it is refilled
        buf ^= 1;
    }

    // epilogue (C layout: VGPR r -> row r (+8 for upper half-lane), col = lane&15)
#pragma unroll
    for (int mi = 0; mi < 2; ++mi)
#pragma unroll
        for (int ni = 0; ni < 4; ++ni)
#pragma unroll
            for (int r = 0; r < 8; ++r) {
                int row = mBase + wm * 32 + mi * 16 + r + half * 8;
                int col = nBase + wn * 64 + ni * 16 + l15;
                float val = acc[mi][ni][r];
                if (OUT_BF16) ((bf16_t*)Cout)[(size_t)row * N + col] = f2bf(val);
                else          ((float*)Cout)[(size_t)row * N + col]  = val;
            }
}

// ---------------------------------------------------------------------------
// Kernel 3: RoPE (fp32 math on bf16 data) + relayout [b,s,h,d] -> [b,h,s,d]
// ---------------------------------------------------------------------------
__global__ __launch_bounds__(256) void rope_kernel(
    const bf16_t* __restrict__ x, bf16_t* __restrict__ out)
{
    long long id = (long long)blockIdx.x * blockDim.x + threadIdx.x;
    if (id >= (long long)MROWS * NHEAD * (HDIM / 2)) return;
    int i  = (int)(id & 63);         // pair index 0..63
    int h  = (int)((id >> 6) & 31);
    int t  = (int)(id >> 11);        // token 0..2047
    int s  = t & (SEQ - 1);
    int b  = t >> 10;

    // inv = theta^(-2i/d) = exp(-ln(1e4) * 2i/128)
    float invf = __expf(-9.210340371976184f * ((float)(2 * i) / 128.0f));
    float ang  = (float)s * invf;
    float sn, cs;
    __sincosf(ang, &sn, &cs);

    const bf16_t* src = x + (size_t)t * HID + h * HDIM;
    float x1 = bf2f(src[i]);
    float x2 = bf2f(src[i + 64]);

    bf16_t* dst = out + (((size_t)(b * NHEAD + h) * SEQ) + s) * HDIM;
    dst[i]      = f2bf(x1 * cs - x2 * sn);
    dst[i + 64] = f2bf(x2 * cs + x1 * sn);
}

// ---------------------------------------------------------------------------
// Kernel 4: V relayout [b,s,h,d] -> Vt [b,h,d,s]
// ---------------------------------------------------------------------------
__global__ __launch_bounds__(256) void reorder_v(
    const bf16_t* __restrict__ v, bf16_t* __restrict__ vt)
{
    long long id = (long long)blockIdx.x * blockDim.x + threadIdx.x;
    if (id >= (long long)BATCH * NHEAD * HDIM * SEQ) return;
    int s = (int)(id & (SEQ - 1));
    int d = (int)((id >> 10) & (HDIM - 1));
    int h = (int)((id >> 17) & (NHEAD - 1));
    int b = (int)(id >> 22);
    vt[id] = v[(size_t)(b * SEQ + s) * HID + h * HDIM + d];
}

// ---------------------------------------------------------------------------
// Kernel 5: causal flash attention with bf16 WMMA.
// One wave per (batch*head, 16-query tile). Keys processed 32 at a time.
// Q,K in [b,h,s,d]; V in [b,h,d,s]; output O fp32 in [b,s,h*d].
// ---------------------------------------------------------------------------
__global__ __launch_bounds__(32) void flash_attn(
    const bf16_t* __restrict__ Q, const bf16_t* __restrict__ Kr,
    const bf16_t* __restrict__ Vt, float* __restrict__ O)
{
    __shared__ bf16_t Plds[16 * 32];

    const int lane = threadIdx.x;
    const int half = lane >> 4;
    const int l15  = lane & 15;
    const int qtile = blockIdx.x;           // 0..63
    const int bh    = blockIdx.y;           // 0..63
    const int qbase = qtile * 16;
    const int b = bh >> 5, h = bh & 31;
    const float sm = 0.08838834764831845f;  // 1/sqrt(128)

    // Q fragments: A-matrix layout, 4 chunks of K(d)=32
    v16bf qf[4];
    const bf16_t* qrow = Q + ((size_t)bh * SEQ + qbase + l15) * HDIM;
#pragma unroll
    for (int dc = 0; dc < 4; ++dc)
        qf[dc] = ld16(qrow + dc * 32 + half * 16);

    v8f oacc[8] = {};
    float mrow[8], lrow[8];
#pragma unroll
    for (int r = 0; r < 8; ++r) { mrow[r] = -3.0e38f; lrow[r] = 0.0f; }

    for (int kv = 0; kv < qbase + 16; kv += 32) {
        // S = Q K^T for this 32-key block (two 16-col tiles)
        v8f sacc[2] = {};
#pragma unroll
        for (int nt = 0; nt < 2; ++nt) {
            const bf16_t* krow = Kr + ((size_t)bh * SEQ + kv + nt * 16 + l15) * HDIM;
#pragma unroll
            for (int dc = 0; dc < 4; ++dc) {
                v16bf kf = ld16(krow + dc * 32 + half * 16);
                sacc[nt] = __builtin_amdgcn_wmma_f32_16x16x32_bf16(
                    false, qf[dc], false, kf, (short)0, sacc[nt], false, false);
            }
        }

        __syncthreads();   // previous P reads done before overwrite
        // scale, causal mask, online softmax stats, stage P into LDS
#pragma unroll
        for (int r = 0; r < 8; ++r) {
            int qrowg = qbase + r + half * 8;
            float s0 = sacc[0][r] * sm;
            float s1 = sacc[1][r] * sm;
            if (kv + l15      > qrowg) s0 = -3.0e38f;
            if (kv + 16 + l15 > qrowg) s1 = -3.0e38f;
            float m = fmaxf(s0, s1);
#pragma unroll
            for (int off = 1; off < 16; off <<= 1)
                m = fmaxf(m, __shfl_xor(m, off));
            float mnew = fmaxf(mrow[r], m);
            float corr = __expf(mrow[r] - mnew);
            float p0 = __expf(s0 - mnew);
            float p1 = __expf(s1 - mnew);
            float rs = p0 + p1;
#pragma unroll
            for (int off = 1; off < 16; off <<= 1)
                rs += __shfl_xor(rs, off);
            lrow[r] = lrow[r] * corr + rs;
            mrow[r] = mnew;
#pragma unroll
            for (int t = 0; t < 8; ++t) oacc[t][r] *= corr;
            Plds[(r + half * 8) * 32 + l15]      = f2bf(p0);
            Plds[(r + half * 8) * 32 + 16 + l15] = f2bf(p1);
        }
        __syncthreads();

        // O += P(16x32) * V(32x128)
        const bf16_t* pbase = Plds + l15 * 32;
        v16bf pf = ld16(pbase + half * 16);
#pragma unroll
        for (int dt = 0; dt < 8; ++dt) {
            const bf16_t* vrow = Vt + ((size_t)bh * HDIM + dt * 16 + l15) * SEQ + kv + half * 16;
            v16bf vf = ld16(vrow);
            oacc[dt] = __builtin_amdgcn_wmma_f32_16x16x32_bf16(
                false, pf, false, vf, (short)0, oacc[dt], false, false);
        }
    }

    // normalize and store O (fp32, [b,s,h*128+d])
#pragma unroll
    for (int dt = 0; dt < 8; ++dt)
#pragma unroll
        for (int r = 0; r < 8; ++r) {
            int q = qbase + r + half * 8;
            int d = dt * 16 + l15;
            O[((size_t)(b * SEQ + q)) * HID + h * HDIM + d] = oacc[dt][r] / lrow[r];
        }
}

// ---------------------------------------------------------------------------
// Launch
// ---------------------------------------------------------------------------
extern "C" void kernel_launch(void* const* d_in, const int* in_sizes, int n_in,
                              void* d_out, int out_size, void* d_ws, size_t ws_size,
                              hipStream_t stream) {
    const float* X  = (const float*)d_in[0];
    const float* Wq = (const float*)d_in[1];
    const float* Wk = (const float*)d_in[2];
    const float* Wv = (const float*)d_in[3];
    const float* Wo = (const float*)d_in[4];

    char* ws = (char*)d_ws;
    size_t off = 0;
    auto alloc = [&](size_t bytes) -> char* {
        char* p = ws + off;
        off = (off + bytes + 255) & ~(size_t)255;
        return p;
    };
    const size_t actB = (size_t)MROWS * HID * sizeof(bf16_t);   // 16 MB
    const size_t wB   = (size_t)HID * HID * sizeof(bf16_t);     // 32 MB

    bf16_t* Xq  = (bf16_t*)alloc(actB);
    bf16_t* Wqb = (bf16_t*)alloc(wB);
    bf16_t* Wkb = (bf16_t*)alloc(wB);
    bf16_t* Wvb = (bf16_t*)alloc(wB);
    bf16_t* Wob = (bf16_t*)alloc(wB);
    bf16_t* Qb  = (bf16_t*)alloc(actB);
    bf16_t* Kb  = (bf16_t*)alloc(actB);
    bf16_t* Vb  = (bf16_t*)alloc(actB);
    bf16_t* Qr  = (bf16_t*)alloc(actB);
    bf16_t* Krr = (bf16_t*)alloc(actB);
    bf16_t* Vt  = (bf16_t*)alloc(actB);
    float*  Of  = (float*)alloc((size_t)MROWS * HID * sizeof(float)); // 32 MB
    bf16_t* Oq  = Xq;  // reuse (Xq dead after QKV projections)

    // 1) fake-quant activations + weights -> bf16
    {
        long long ga = (long long)MROWS * GROUPS_PER_ROW;
        long long gw = (long long)HID * GROUPS_PER_ROW;
        fakequant_kernel<<<(unsigned)((ga + 255) / 256), 256, 0, stream>>>(X,  Xq,  ga);
        fakequant_kernel<<<(unsigned)((gw + 255) / 256), 256, 0, stream>>>(Wq, Wqb, gw);
        fakequant_kernel<<<(unsigned)((gw + 255) / 256), 256, 0, stream>>>(Wk, Wkb, gw);
        fakequant_kernel<<<(unsigned)((gw + 255) / 256), 256, 0, stream>>>(Wv, Wvb, gw);
        fakequant_kernel<<<(unsigned)((gw + 255) / 256), 256, 0, stream>>>(Wo, Wob, gw);
    }

    // 2) QKV projections (bf16 WMMA, async-LDS double buffered)
    dim3 gg(HID / 128, MROWS / 128);
    gemm_bf16<true><<<gg, 256, 0, stream>>>(Xq, Wqb, Qb, MROWS, HID, HID);
    gemm_bf16<true><<<gg, 256, 0, stream>>>(Xq, Wkb, Kb, MROWS, HID, HID);
    gemm_bf16<true><<<gg, 256, 0, stream>>>(Xq, Wvb, Vb, MROWS, HID, HID);

    // 3) RoPE + relayouts
    {
        long long nr = (long long)MROWS * NHEAD * (HDIM / 2);
        rope_kernel<<<(unsigned)((nr + 255) / 256), 256, 0, stream>>>(Qb, Qr);
        rope_kernel<<<(unsigned)((nr + 255) / 256), 256, 0, stream>>>(Kb, Krr);
        long long nv = (long long)BATCH * NHEAD * HDIM * SEQ;
        reorder_v<<<(unsigned)((nv + 255) / 256), 256, 0, stream>>>(Vb, Vt);
    }

    // 4) causal flash attention (WMMA), fp32 O in [b,s,h*d]
    {
        dim3 ga(SEQ / 16, BATCH * NHEAD);
        flash_attn<<<ga, 32, 0, stream>>>(Qr, Krr, Vt, Of);
    }

    // 5) fake-quant O, final projection straight into d_out (fp32)
    {
        long long go = (long long)MROWS * GROUPS_PER_ROW;
        fakequant_kernel<<<(unsigned)((go + 255) / 256), 256, 0, stream>>>(Of, Oq, go);
        gemm_bf16<false><<<gg, 256, 0, stream>>>(Oq, Wob, d_out, MROWS, HID, HID);
    }
}
